// ChunkHead_21337397527049
// MI455X (gfx1250) — matrix operand order repaired and checked by reading, not compiled, not run on hardware
//
#include <hip/hip_runtime.h>

// ---------------------------------------------------------------------------
// ChunkHead for MI455X (gfx1250, wave32).  All GEMM-class math runs on
// v_wmma_f32_16x16x32_bf16 with f32 accumulation; softmax / loss / argmax
// stay in f32 VALU.  Problem: B=4, S=4096, D=DC=1024.
//
// v2: GEMM stages the shared weight tile through LDS (double-buffered,
//     8 waves reuse one 64x64 bf16 chunk), register double-buffering around
//     every WMMA so waits are partial, attention re-tiled to 8 waves x
//     16x128 accumulators for lower VGPR pressure / better occupancy,
//     global_prefetch for the next A chunk.
// ---------------------------------------------------------------------------

typedef __attribute__((ext_vector_type(8)))  float          v8f;
typedef __attribute__((ext_vector_type(16))) __bf16         v16bf;
typedef __attribute__((ext_vector_type(8)))  unsigned short v8u;
typedef __attribute__((ext_vector_type(16))) unsigned short v16u;

static __device__ __forceinline__ unsigned short f2bf(float f) {
  unsigned u = __builtin_bit_cast(unsigned, f);
  unsigned r = u + 0x7FFFu + ((u >> 16) & 1u);   // round-to-nearest-even
  return (unsigned short)(r >> 16);
}

static __device__ __forceinline__ v16bf combine16(v8u lo, v8u hi) {
  v16u r;
#pragma unroll
  for (int i = 0; i < 8; ++i) { r[i] = lo[i]; r[i + 8] = hi[i]; }
  return __builtin_bit_cast(v16bf, r);
}

// A-fragment (16x32 bf16, M rows over lanes 0..15): element e -> K = half*16+e
static __device__ __forceinline__ v16bf load_fragA(const unsigned short* __restrict__ base,
                                                   int ld, int line, int k0, int lane) {
  const int half = lane >> 4;
  const unsigned short* p = base + (size_t)line * ld + k0 + half * 16;
  v8u lo = *(const v8u*)p;          // K = half*16 .. +7
  v8u hi = *(const v8u*)(p + 8);    // K = half*16+8 .. +7
  return combine16(lo, hi);
}

// B-fragment (32x16 bf16, N cols over lanes 0..15): element e -> K = half*16+e
// `line` is the row of an [N][K]-major array holding that column's K vector.
static __device__ __forceinline__ v16bf load_fragB(const unsigned short* __restrict__ base,
                                                   int ld, int line, int k0, int lane) {
  const int half = lane >> 4;
  const unsigned short* p = base + (size_t)line * ld + k0 + half * 16;
  v8u lo = *(const v8u*)p;
  v8u hi = *(const v8u*)(p + 8);
  return combine16(lo, hi);
}

static __device__ __forceinline__ v8f wmma_bf16(v16bf a, v16bf b, v8f c) {
  return __builtin_amdgcn_wmma_f32_16x16x32_bf16(false, a, false, b, (short)0, c,
                                                 false, false);
}

// ---------------------------------------------------------------------------
// 1) logits / loss / argmax:  one wave per token, D=1024 dot with Wl [D,4].
// ---------------------------------------------------------------------------
__global__ void loss_pred_kernel(const float* __restrict__ h, const int* __restrict__ lab,
                                 const float* __restrict__ Wl, const float* __restrict__ bl,
                                 int* __restrict__ pred, float* __restrict__ lossOut,
                                 int BS, int D, float invN) {
  int wid  = blockIdx.x * (blockDim.x >> 5) + (threadIdx.x >> 5);
  int lane = threadIdx.x & 31;
  if (wid >= BS) return;
  const float* hp = h + (size_t)wid * D;
  float a0 = 0.f, a1 = 0.f, a2 = 0.f, a3 = 0.f;
  for (int i = lane; i < D; i += 32) {
    float hv = hp[i];
    float4 w = *(const float4*)(Wl + i * 4);
    a0 += hv * w.x; a1 += hv * w.y; a2 += hv * w.z; a3 += hv * w.w;
  }
#pragma unroll
  for (int off = 16; off > 0; off >>= 1) {
    a0 += __shfl_xor(a0, off);
    a1 += __shfl_xor(a1, off);
    a2 += __shfl_xor(a2, off);
    a3 += __shfl_xor(a3, off);
  }
  if (lane == 0) {
    float l0 = a0 + bl[0], l1 = a1 + bl[1], l2 = a2 + bl[2], l3 = a3 + bl[3];
    int am = 0; float mx = l0;
    if (l1 > mx) { mx = l1; am = 1; }
    if (l2 > mx) { mx = l2; am = 2; }
    if (l3 > mx) { mx = l3; am = 3; }
    float lse = mx + __logf(__expf(l0 - mx) + __expf(l1 - mx) +
                            __expf(l2 - mx) + __expf(l3 - mx));
    int lv = lab[wid];
    float ll = (lv == 0) ? l0 : (lv == 1) ? l1 : (lv == 2) ? l2 : l3;
    atomicAdd(lossOut, (lse - ll) * invN);
    pred[wid] = am;
  }
}

// ---------------------------------------------------------------------------
// 2) BIOS chunk scan (sequential FSM, 1 thread/batch).
// ---------------------------------------------------------------------------
__global__ void chunk_kernel(const int* __restrict__ pred, int* __restrict__ cs,
                             int* __restrict__ nC, int B, int S) {
  int b = blockIdx.x * blockDim.x + threadIdx.x;
  if (b >= B) return;
  const int* p = pred + (size_t)b * S;
  int* csb = cs + (size_t)b * (S + 1);
  int C = 0; bool ext = false;
  for (int s = 0; s < S; ++s) {
    int l = p[s];
    bool cont = (l == 1) && ext;
    ext = (l == 0) || cont;
    if (!cont) csb[C++] = s;
  }
  csb[C] = S;
  nC[b] = C;
}

// ---------------------------------------------------------------------------
// 3) Segment-mean pooling -> chunk_emb bf16 [B,S,D] (zero past chunk count).
// ---------------------------------------------------------------------------
__global__ void pool_kernel(const float* __restrict__ h, const int* __restrict__ cs,
                            const int* __restrict__ nC, unsigned short* __restrict__ ce,
                            int S, int D) {
  int b = blockIdx.x / S, c = blockIdx.x % S;
  unsigned short* out = ce + ((size_t)(b * S + c)) * D;
  if (c >= nC[b]) {
    for (int d = threadIdx.x; d < D; d += blockDim.x) out[d] = 0;
    return;
  }
  const int* csb = cs + (size_t)b * (S + 1);
  int s0 = csb[c], s1 = csb[c + 1];
  float inv = 1.f / (float)(s1 - s0);
  const float* hb = h + ((size_t)b * S) * D;
  for (int d = threadIdx.x; d < D; d += blockDim.x) {
    float sum = 0.f;
    for (int s = s0; s < s1; ++s) sum += hb[(size_t)s * D + d];
    out[d] = f2bf(sum * inv);
  }
}

// ---------------------------------------------------------------------------
// Elementwise f32 -> bf16 and transposed weight pack W[K,N] -> WT[N,K] bf16.
// ---------------------------------------------------------------------------
__global__ void cvt_bf16_kernel(const float* __restrict__ in,
                                unsigned short* __restrict__ out, int n) {
  int i = blockIdx.x * blockDim.x + threadIdx.x;
  if (i < n) out[i] = f2bf(in[i]);
}

__global__ void pack_wT_kernel(const float* __restrict__ W,
                               unsigned short* __restrict__ WT, int K, int N) {
  int i = blockIdx.x * blockDim.x + threadIdx.x;
  if (i >= K * N) return;
  int k = i / N, n = i - k * N;
  WT[(size_t)n * K + k] = f2bf(W[i]);
}

__global__ void init_loss_kernel(float* p) {
  if (threadIdx.x == 0 && blockIdx.x == 0) *p = 0.f;
}

// ---------------------------------------------------------------------------
// 4) WMMA GEMM with LDS-staged weights.
//    Block = 256 threads (8 waves) -> 128x64 output tile.  The shared 64x64
//    bf16 chunk of BT is staged global->reg->LDS double-buffered; all 8
//    waves consume it via ds_load_b128 (8x reuse).  A fragments come from
//    global (row-unique per wave), next chunk prefetched.
//    MODE 0: f32 row-major; MODE 1: bf16 row-major;
//    MODE 2: bf16 transposed per batch ([b][n][m_local], ld = Spad).
// ---------------------------------------------------------------------------
#define BSTRIDE 72   // 64 + 8 ushorts padding: 144B lines, 16B aligned

template <int MODE>
__global__ __launch_bounds__(256)
void gemm_bf16(const unsigned short* __restrict__ A,
               const unsigned short* __restrict__ BT,
               const float* __restrict__ bias, void* __restrict__ Out,
               int M, int N, int K, const int* __restrict__ nC, int Spad) {
  __shared__ unsigned short sB[2][64 * BSTRIDE];

  const int nBlocks = N >> 6;
  const int mBlock = blockIdx.x / nBlocks;
  const int nBlock = blockIdx.x - mBlock * nBlocks;
  const int m0blk = mBlock << 7;     // 128 rows per block
  const int n0 = nBlock << 6;        // 64 cols per block
  int b = 0, localM0 = m0blk;
  if (nC) {                           // uniform early-exit (before barriers)
    b = m0blk / Spad;
    localM0 = m0blk - b * Spad;
    if (localM0 >= nC[b]) return;
  }
  const int tid = threadIdx.x;
  const int wv = tid >> 5, lane = tid & 31;
  const int col = lane & 15, half = lane >> 4;
  const int rowA = m0blk + (wv << 4) + col;
  const unsigned short* aRow = A + (size_t)rowA * K;

  // staging assignment: 4 threads per line, 16 ushorts (32B) each
  const int sLine = tid >> 2;
  const int sSub  = (tid & 3) << 4;
  const unsigned short* gB = BT + (size_t)(n0 + sLine) * K + sSub;
  unsigned short* lB0 = &sB[0][sLine * BSTRIDE + sSub];
  unsigned short* lB1 = &sB[1][sLine * BSTRIDE + sSub];

  { // stage chunk 0
    v8u t0 = *(const v8u*)gB;
    v8u t1 = *(const v8u*)(gB + 8);
    *(v8u*)lB0 = t0; *(v8u*)(lB0 + 8) = t1;
  }
  __syncthreads();

  v8f z = {0.f, 0.f, 0.f, 0.f, 0.f, 0.f, 0.f, 0.f};
  v8f acc[4] = {z, z, z, z};
  int cur = 0;
  for (int kk = 0; kk < K; kk += 64) {
    const bool nxt = (kk + 64) < K;
    v8u t0, t1;
    if (nxt) {                       // issue next-chunk global loads early
      const unsigned short* g = gB + kk + 64;
      t0 = *(const v8u*)g; t1 = *(const v8u*)(g + 8);
      __builtin_prefetch(aRow + kk + 128, 0, 1);   // global_prefetch_b8
    }
    const unsigned short* bufc = sB[cur];
#pragma unroll
    for (int k2 = 0; k2 < 64; k2 += 32) {
      v16bf a = load_fragA(A, K, rowA, kk + k2, lane);
#pragma unroll
      for (int f = 0; f < 4; ++f) {
        const unsigned short* p = bufc + (f * 16 + col) * BSTRIDE + k2 + half * 16;
        v8u lo = *(const v8u*)p;
        v8u hi = *(const v8u*)(p + 8);
        acc[f] = wmma_bf16(a, combine16(lo, hi), acc[f]);
      }
    }
    if (nxt) {                       // park staged regs into the other buffer
      unsigned short* l = cur ? lB0 : lB1;
      *(v8u*)l = t0; *(v8u*)(l + 8) = t1;
    }
    __syncthreads();
    cur ^= 1;
  }

#pragma unroll
  for (int f = 0; f < 4; ++f) {
    int n = n0 + f * 16 + col;
    float bv = bias ? bias[n] : 0.f;
#pragma unroll
    for (int j = 0; j < 8; ++j) {
      float v = acc[f][j] + bv;
      int row = j + 8 * half;
      int mg = m0blk + (wv << 4) + row;
      if (MODE == 0)
        ((float*)Out)[(size_t)mg * N + n] = v;
      else if (MODE == 1)
        ((unsigned short*)Out)[(size_t)mg * N + n] = f2bf(v);
      else
        ((unsigned short*)Out)[((size_t)(b * N + n)) * Spad +
                               (localM0 + (wv << 4) + row)] = f2bf(v);
    }
  }
}

// ---------------------------------------------------------------------------
// 5) Flash-style chunk attention.  Block = 8 waves (256 thr), one 16-row
//    q tile.  Waves split K=1024 for q@k^T partials (LDS reduce), online
//    softmax in f32 registers, P rebuilt as bf16 A-fragment, PV via WMMA
//    into 16x128-per-wave f32 accumulators (64 VGPRs -> decent occupancy).
//    v stored transposed ([b][d][c]) so PV B-fragments are contiguous.
// ---------------------------------------------------------------------------
__global__ __launch_bounds__(256)
void attn_fa_kernel(const unsigned short* __restrict__ qb,   // [B*S, DC]
                    const unsigned short* __restrict__ kb,   // [B*S, DC]
                    const unsigned short* __restrict__ vT,   // [B, D, S]
                    const int* __restrict__ nC,
                    unsigned short* __restrict__ attOut,     // [B*S, D]
                    int S, int D, int DC, float scale) {
  __shared__ float lds_part[8][16][32];
  __shared__ float lds_red[16][32];
  __shared__ float lds_m[16];

  const int tilesPerB = S >> 4;
  const int b  = blockIdx.x / tilesPerB;
  const int m0 = (blockIdx.x - b * tilesPerB) << 4;
  const int tid = threadIdx.x, lane = tid & 31, wid = tid >> 5;
  const int col = lane & 15, half = lane >> 4;
  const int Cb = nC[b];
  const int Cround = (Cb + 31) & ~31;

  v8f z = {0.f, 0.f, 0.f, 0.f, 0.f, 0.f, 0.f, 0.f};
  v8f acc[8];
#pragma unroll
  for (int f = 0; f < 8; ++f) acc[f] = z;
  float mReg[8], lReg[8];
#pragma unroll
  for (int j = 0; j < 8; ++j) { mReg[j] = -1e30f; lReg[j] = 0.f; }

  const int qRow  = b * S + m0 + col;   // A-fragment line for q
  const int kBase = wid << 7;           // this wave's 128-wide K slice
  const int nBase = wid << 7;           // this wave's 128-wide D slice
  const unsigned short* vTb = vT + (size_t)b * D * S;

  for (int c0 = 0; c0 < Cround; c0 += 32) {
    // ---- partial scores over our K slice (reg double-buffered) -----------
    v8f s0 = z, s1 = z;
    {
      v16bf qa  = load_fragA(qb, DC, qRow, kBase, lane);
      v16bf k0f = load_fragB(kb, DC, b * S + c0 + col, kBase, lane);
      v16bf k1f = load_fragB(kb, DC, b * S + c0 + 16 + col, kBase, lane);
#pragma unroll
      for (int it = 0; it < 4; ++it) {
        v16bf qa_n = qa, k0_n = k0f, k1_n = k1f;
        if (it < 3) {
          int kq = kBase + (it + 1) * 32;
          qa_n = load_fragA(qb, DC, qRow, kq, lane);
          k0_n = load_fragB(kb, DC, b * S + c0 + col, kq, lane);
          k1_n = load_fragB(kb, DC, b * S + c0 + 16 + col, kq, lane);
        }
        s0 = wmma_bf16(qa, k0f, s0);
        s1 = wmma_bf16(qa, k1f, s1);
        qa = qa_n; k0f = k0_n; k1f = k1_n;
      }
    }
#pragma unroll
    for (int j = 0; j < 8; ++j) {
      int r = j + 8 * half;
      lds_part[wid][r][col]      = s0[j];
      lds_part[wid][r][col + 16] = s1[j];
    }
    __syncthreads();
    // ---- reduce 8 wave partials -> full 16x32 score tile -----------------
    for (int i = tid; i < 512; i += 256) {
      int r = i >> 5, c = i & 31;
      float v = 0.f;
#pragma unroll
      for (int w = 0; w < 8; ++w) v += lds_part[w][r][c];
      lds_red[r][c] = v;
    }
    __syncthreads();
    // ---- online softmax stats (each lane owns rows j + 8*half) -----------
    int cv = Cb - c0; if (cv > 32) cv = 32;
    float corr[8];
#pragma unroll
    for (int j = 0; j < 8; ++j) {
      int r = j + 8 * half;
      float mN = mReg[j];
      for (int c = 0; c < cv; ++c) mN = fmaxf(mN, lds_red[r][c] * scale);
      float co = __expf(mReg[j] - mN);
      float ls = 0.f;
      for (int c = 0; c < cv; ++c) ls += __expf(lds_red[r][c] * scale - mN);
      lReg[j] = lReg[j] * co + ls;
      mReg[j] = mN;
      corr[j] = co;
    }
    if (wid == 0 && col == 0) {
#pragma unroll
      for (int j = 0; j < 8; ++j) lds_m[j + 8 * half] = mReg[j];
    }
#pragma unroll
    for (int f = 0; f < 8; ++f)
#pragma unroll
      for (int j = 0; j < 8; ++j) acc[f][j] *= corr[j];
    __syncthreads();
    // ---- build P as a bf16 A-fragment (masked past Cb) -------------------
    v16u praw;
#pragma unroll
    for (int e = 0; e < 16; ++e) {
      int kk = half * 16 + e;
      float p = (kk < cv) ? __expf(lds_red[col][kk] * scale - lds_m[col]) : 0.f;
      praw[e] = f2bf(p);
    }
    v16bf pf = __builtin_bit_cast(v16bf, praw);
    // ---- PV: accumulate into the wave's 16x128 output slice --------------
    {
      v16bf vf = load_fragB(vTb, S, nBase + col, c0, lane);
#pragma unroll
      for (int f = 0; f < 8; ++f) {
        v16bf vfn = vf;
        if (f < 7) vfn = load_fragB(vTb, S, nBase + (f + 1) * 16 + col, c0, lane);
        acc[f] = wmma_bf16(pf, vf, acc[f]);
        vf = vfn;
      }
    }
  }
  // ---- normalize + store attended (bf16, row-major) ----------------------
#pragma unroll
  for (int f = 0; f < 8; ++f) {
    int n = nBase + f * 16 + col;
#pragma unroll
    for (int j = 0; j < 8; ++j) {
      int row = j + 8 * half;
      float l = lReg[j];
      float ov = acc[f][j] / (l > 0.f ? l : 1.f);
      attOut[((size_t)(b * S + m0 + row)) * D + n] = f2bf(ov);
    }
  }
}

// ---------------------------------------------------------------------------
// Host-side orchestration.
// ---------------------------------------------------------------------------
extern "C" void kernel_launch(void* const* d_in, const int* in_sizes, int n_in,
                              void* d_out, int out_size, void* d_ws, size_t ws_size,
                              hipStream_t stream) {
  const int B = 4, S = 4096, D = 1024, DC = 1024;
  const int BS = B * S;

  const float* h   = (const float*)d_in[0];
  const int*   pl  = (const int*)d_in[1];
  const float* Wl  = (const float*)d_in[2];
  const float* bl  = (const float*)d_in[3];
  const float* Wq  = (const float*)d_in[4];
  const float* bq  = (const float*)d_in[5];
  const float* Wk  = (const float*)d_in[6];
  const float* bk  = (const float*)d_in[7];
  const float* Wv  = (const float*)d_in[8];
  const float* bv  = (const float*)d_in[9];
  const float* Wo  = (const float*)d_in[10];
  const float* bo  = (const float*)d_in[11];

  float* out   = (float*)d_out;            // [B*S*D] ch_out
  float* lossP = out + (size_t)BS * D;     // [1] loss

  char* ws = (char*)d_ws;
  auto alloc = [&](size_t bytes) -> void* {
    void* p = (void*)ws;
    ws += (bytes + 255) & ~(size_t)255;
    return p;
  };
  unsigned short* h_bf  = (unsigned short*)alloc((size_t)BS * D * 2);
  unsigned short* ce_bf = (unsigned short*)alloc((size_t)BS * D * 2);
  unsigned short* q_bf  = (unsigned short*)alloc((size_t)BS * DC * 2);
  unsigned short* k_bf  = (unsigned short*)alloc((size_t)BS * DC * 2);
  unsigned short* vT_bf = (unsigned short*)alloc((size_t)B * D * S * 2);
  unsigned short* att_bf= (unsigned short*)alloc((size_t)BS * D * 2);
  unsigned short* WqT   = (unsigned short*)alloc((size_t)D * DC * 2);
  unsigned short* WkT   = (unsigned short*)alloc((size_t)D * DC * 2);
  unsigned short* WvT   = (unsigned short*)alloc((size_t)D * D * 2);
  unsigned short* WoT   = (unsigned short*)alloc((size_t)D * D * 2);
  int* pred = (int*)alloc((size_t)BS * 4);
  int* cs   = (int*)alloc((size_t)B * (S + 1) * 4);
  int* nC   = (int*)alloc((size_t)B * 4);

  init_loss_kernel<<<1, 32, 0, stream>>>(lossP);

  // 1) logits / loss / argmax
  loss_pred_kernel<<<BS / 8, 256, 0, stream>>>(h, pl, Wl, bl, pred, lossP,
                                               BS, D, 1.0f / (float)BS);
  // 2) BIOS scan
  chunk_kernel<<<1, 32, 0, stream>>>(pred, cs, nC, B, S);
  // 3) pooling -> chunk_emb bf16 (zero-filled past chunk count)
  pool_kernel<<<BS, 256, 0, stream>>>(h, cs, nC, ce_bf, S, D);
  // 4) bf16 conversion + transposed weight packs
  cvt_bf16_kernel<<<(BS * D) / 256, 256, 0, stream>>>(h, h_bf, BS * D);
  pack_wT_kernel<<<(D * DC) / 256, 256, 0, stream>>>(Wq, WqT, D, DC);
  pack_wT_kernel<<<(D * DC) / 256, 256, 0, stream>>>(Wk, WkT, D, DC);
  pack_wT_kernel<<<(D * D) / 256, 256, 0, stream>>>(Wv, WvT, D, D);
  pack_wT_kernel<<<(D * D) / 256, 256, 0, stream>>>(Wo, WoT, D, D);

  // 5) projections: block = 128x64 tile, LDS-staged weights
  auto gemm_blocks = [](int M, int N) { return (M / 128) * (N / 64); };
  gemm_bf16<1><<<gemm_blocks(BS, DC), 256, 0, stream>>>(h_bf, WqT, bq, q_bf,
                                                        BS, DC, D, nullptr, 0);
  gemm_bf16<1><<<gemm_blocks(BS, DC), 256, 0, stream>>>(ce_bf, WkT, bk, k_bf,
                                                        BS, DC, D, nC, S);
  gemm_bf16<2><<<gemm_blocks(BS, D), 256, 0, stream>>>(ce_bf, WvT, bv, vT_bf,
                                                       BS, D, D, nC, S);
  // 6) fused masked-softmax attention
  attn_fa_kernel<<<B * (S / 16), 256, 0, stream>>>(q_bf, k_bf, vT_bf, nC, att_bf,
                                                   S, D, DC,
                                                   1.0f / 32.0f /* 1/sqrt(DC) */);
  // 7) output projection straight into d_out (f32)
  gemm_bf16<0><<<gemm_blocks(BS, D), 256, 0, stream>>>(att_bf, WoT, bo, (void*)out,
                                                       BS, D, D, nullptr, 0);
}
